// ExpertParallelMoELayer_9990093930652
// MI455X (gfx1250) — matrix operand order repaired and checked by reading, not compiled, not run on hardware
//
#include <hip/hip_runtime.h>
#include <hip/hip_bf16.h>

// ---------- vector types (plain clang ext vectors: safe in unions) ----------
typedef __attribute__((ext_vector_type(16))) __bf16         v16bf;
typedef __attribute__((ext_vector_type(8)))  float          v8f;
typedef __attribute__((ext_vector_type(2)))  unsigned int   u32x2;
typedef __attribute__((ext_vector_type(4)))  unsigned int   u32x4;
typedef __attribute__((ext_vector_type(4)))  unsigned short u16x4;
typedef __attribute__((ext_vector_type(4)))  float          f32x4;

__device__ __forceinline__ unsigned short f2bf(float f) {
    return __builtin_bit_cast(unsigned short, (__bf16)f);
}

union FragU {
    v16bf v;
    u32x2 d2[4];
    u32x4 d4[2];
};

// A fragment: 16x32 bf16 (MxK) slice of an LDS tile with row stride ldk.
// Contiguous k order used here (matches load_frag_b): lanes 0-15 -> M=lane, K in {0..15}; lanes 16-31 -> M=lane-16,
// K in {16..31}.
__device__ __forceinline__ v16bf load_frag_a(const unsigned short* tile, int ldk, int lane) {
    const int m  = lane & 15;
    const int kb = (lane & 16) ? 16 : 0;
    const unsigned short* row = tile + m * ldk;
    FragU f;
    f.d2[0] = *(const u32x2*)(row + kb + 0);   // VGPR0,1 : K kb+0..3
    f.d2[1] = *(const u32x2*)(row + kb + 4);   // VGPR2,3 : K kb+4..7
    f.d2[2] = *(const u32x2*)(row + kb + 8);   // VGPR4,5 : K kb+8..11
    f.d2[3] = *(const u32x2*)(row + kb + 12);  // VGPR6,7 : K kb+12..15
    return f.v;
}

// B fragment: 32x16 bf16 (KxN); weight tile stored n-major ([n][k], stride ldk).
// Lanes 0-15 -> column n=lane, K=0..15 in VGPR0..7; lanes 16-31 -> K=16..31.
__device__ __forceinline__ v16bf load_frag_b(const unsigned short* tile, int ldk, int lane) {
    const int n  = lane & 15;
    const int kb = (lane & 16) ? 16 : 0;
    const unsigned short* row = tile + n * ldk;
    FragU f;
    f.d4[0] = *(const u32x4*)(row + kb + 0);   // VGPR0..3 : K kb..kb+7
    f.d4[1] = *(const u32x4*)(row + kb + 8);   // VGPR4..7 : K kb+8..kb+15
    return f.v;
}

__device__ __forceinline__ v8f wmma_bf16(v16bf a, v16bf b, v8f c) {
    return __builtin_amdgcn_wmma_f32_16x16x32_bf16(
        false, a, false, b, (short)0, c, false, false);
}

// ---------------- CDNA5 async global->LDS (ASYNCcnt) ------------------------
// LDS address comes from the VGPR operand; flat LDS aperture keeps the LDS offset in the low
// 32 bits of the generic pointer.
__device__ __forceinline__ void async_ld_b128(const unsigned short* g, const unsigned short* l) {
    const unsigned lds_off = (unsigned)(unsigned long long)l;
    asm volatile("global_load_async_to_lds_b128 %0, %1, off"
                 :: "v"(lds_off), "v"(g)
                 : "memory");
}
__device__ __forceinline__ void wait_async0() { asm volatile("s_wait_asynccnt 0x0" ::: "memory"); }
__device__ __forceinline__ void wait_async8() { asm volatile("s_wait_asynccnt 0x8" ::: "memory"); }

// ---------------------------------------------------------------------------
// fp32 -> bf16 bulk conversion (n divisible by 4)
// ---------------------------------------------------------------------------
__global__ __launch_bounds__(256)
void cvt_f32_to_bf16(const float* __restrict__ in, unsigned short* __restrict__ out,
                     long long n) {
    long long i = ((long long)blockIdx.x * 256 + threadIdx.x) * 4;
    if (i < n) {
        f32x4 f = *(const f32x4*)(in + i);
        u16x4 o;
        o.x = f2bf(f.x); o.y = f2bf(f.y); o.z = f2bf(f.z); o.w = f2bf(f.w);
        *(u16x4*)(out + i) = o;
    }
}

// ---------------------------------------------------------------------------
// GEMM1 fused: Hb = bf16( silu(X @ W1^T) * (X @ W3^T) )
// X:[N,H] bf16, W1/W3:[I,H] bf16 (NT, K=H), Hb:[N,I] bf16
// Block tile 128(M) x 64(I), BK=64, double-buffered LDS, async-to-LDS staging.
// 8 waves: 4 along M x 2 along N; each wave: 32x32 for BOTH matmuls.
// ---------------------------------------------------------------------------
__global__ __launch_bounds__(256)
void moe_gemm13_swiglu(const unsigned short* __restrict__ Xb,
                       const unsigned short* __restrict__ W1b,
                       const unsigned short* __restrict__ W3b,
                       unsigned short* __restrict__ Hb,
                       int N, int H, int I) {
    __shared__ __align__(16) unsigned short sA [2][128 * 64];
    __shared__ __align__(16) unsigned short sB1[2][64 * 64];
    __shared__ __align__(16) unsigned short sB3[2][64 * 64];

    const int tid  = threadIdx.x;
    const int lane = tid & 31;
    const int wave = tid >> 5;
    const int wm   = wave & 3;   // 4 waves * 32 rows = 128
    const int wn   = wave >> 2;  // 2 waves * 32 cols = 64

    const int m0 = blockIdx.y * 128;
    const int n0 = blockIdx.x * 64;

    const int lr = tid >> 3;         // 0..31
    const int lc = (tid & 7) * 8;    // 16B chunks across K=64

    v8f acc1[2][2], acc3[2][2];
#pragma unroll
    for (int i = 0; i < 2; ++i)
#pragma unroll
        for (int j = 0; j < 2; ++j) {
            acc1[i][j] = (v8f){0.f,0.f,0.f,0.f,0.f,0.f,0.f,0.f};
            acc3[i][j] = (v8f){0.f,0.f,0.f,0.f,0.f,0.f,0.f,0.f};
        }

    // 8 async b128 loads per thread per tile
    auto issue_tile = [&](int kt, int buf) {
        const int k0 = kt * 64;
#pragma unroll
        for (int i = 0; i < 4; ++i) {
            const int r = lr + i * 32;
            async_ld_b128(Xb + (size_t)(m0 + r) * H + k0 + lc, &sA[buf][r * 64 + lc]);
        }
#pragma unroll
        for (int i = 0; i < 2; ++i) {
            const int r = lr + i * 32;
            async_ld_b128(W1b + (size_t)(n0 + r) * H + k0 + lc, &sB1[buf][r * 64 + lc]);
            async_ld_b128(W3b + (size_t)(n0 + r) * H + k0 + lc, &sB3[buf][r * 64 + lc]);
        }
    };

    const int ksteps = H >> 6;
    issue_tile(0, 0);
    for (int kt = 0; kt < ksteps; ++kt) {
        const int cur = kt & 1;
        if (kt + 1 < ksteps) { issue_tile(kt + 1, cur ^ 1); wait_async8(); }
        else                 { wait_async0(); }
        __syncthreads();

#pragma unroll
        for (int kk = 0; kk < 2; ++kk) {
            v16bf fa[2], fb1[2], fb3[2];
#pragma unroll
            for (int tm = 0; tm < 2; ++tm)
                fa[tm] = load_frag_a(&sA[cur][(wm * 32 + tm * 16) * 64 + kk * 32], 64, lane);
#pragma unroll
            for (int tn = 0; tn < 2; ++tn) {
                fb1[tn] = load_frag_b(&sB1[cur][(wn * 32 + tn * 16) * 64 + kk * 32], 64, lane);
                fb3[tn] = load_frag_b(&sB3[cur][(wn * 32 + tn * 16) * 64 + kk * 32], 64, lane);
            }
#pragma unroll
            for (int tm = 0; tm < 2; ++tm)
#pragma unroll
                for (int tn = 0; tn < 2; ++tn) {
                    acc1[tm][tn] = wmma_bf16(fa[tm], fb1[tn], acc1[tm][tn]);
                    acc3[tm][tn] = wmma_bf16(fa[tm], fb3[tn], acc3[tm][tn]);
                }
        }
        __syncthreads();
    }

    // epilogue: silu(x1) * x3 -> bf16
    const int cn = lane & 15;
    const int mh = (lane >> 4) * 8;  // C layout: lanes 16-31 hold M = r + 8
#pragma unroll
    for (int tm = 0; tm < 2; ++tm)
#pragma unroll
        for (int tn = 0; tn < 2; ++tn) {
            const int r0 = m0 + wm * 32 + tm * 16 + mh;
            const int c  = n0 + wn * 32 + tn * 16 + cn;
#pragma unroll
            for (int r = 0; r < 8; ++r) {
                const float x1 = acc1[tm][tn][r];
                const float x3 = acc3[tm][tn][r];
                const float h  = (x1 / (1.0f + __expf(-x1))) * x3;
                Hb[(size_t)(r0 + r) * I + c] = f2bf(h);
            }
        }
}

// ---------------------------------------------------------------------------
// GEMM2: Out = Hb @ W2^T.  Hb:[N,I] bf16, W2:[H,I] bf16 (NT, K=I), Out fp32.
// Block tile 128(M) x 128(H), BK=64, double-buffered, async-to-LDS staging.
// 8 waves: 4 along M x 2 along N; each wave 32x64 (2x4 WMMA tiles).
// ---------------------------------------------------------------------------
__global__ __launch_bounds__(256)
void moe_gemm2(const unsigned short* __restrict__ Hb,
               const unsigned short* __restrict__ W2b,
               float* __restrict__ Out,
               int N, int H, int I) {
    __shared__ __align__(16) unsigned short sA[2][128 * 64];
    __shared__ __align__(16) unsigned short sB[2][128 * 64];

    const int tid  = threadIdx.x;
    const int lane = tid & 31;
    const int wave = tid >> 5;
    const int wm   = wave & 3;   // 4 * 32 = 128 rows
    const int wn   = wave >> 2;  // 2 * 64 = 128 cols

    const int m0 = blockIdx.y * 128;
    const int n0 = blockIdx.x * 128;

    const int lr = tid >> 3;
    const int lc = (tid & 7) * 8;

    v8f acc[2][4];
#pragma unroll
    for (int i = 0; i < 2; ++i)
#pragma unroll
        for (int j = 0; j < 4; ++j)
            acc[i][j] = (v8f){0.f,0.f,0.f,0.f,0.f,0.f,0.f,0.f};

    auto issue_tile = [&](int kt, int buf) {
        const int k0 = kt * 64;
#pragma unroll
        for (int i = 0; i < 4; ++i) {
            const int r = lr + i * 32;
            async_ld_b128(Hb  + (size_t)(m0 + r) * I + k0 + lc, &sA[buf][r * 64 + lc]);
            async_ld_b128(W2b + (size_t)(n0 + r) * I + k0 + lc, &sB[buf][r * 64 + lc]);
        }
    };

    const int ksteps = I >> 6;
    issue_tile(0, 0);
    for (int kt = 0; kt < ksteps; ++kt) {
        const int cur = kt & 1;
        if (kt + 1 < ksteps) { issue_tile(kt + 1, cur ^ 1); wait_async8(); }
        else                 { wait_async0(); }
        __syncthreads();

#pragma unroll
        for (int kk = 0; kk < 2; ++kk) {
            v16bf fa[2], fb[4];
#pragma unroll
            for (int tm = 0; tm < 2; ++tm)
                fa[tm] = load_frag_a(&sA[cur][(wm * 32 + tm * 16) * 64 + kk * 32], 64, lane);
#pragma unroll
            for (int tn = 0; tn < 4; ++tn)
                fb[tn] = load_frag_b(&sB[cur][(wn * 64 + tn * 16) * 64 + kk * 32], 64, lane);
#pragma unroll
            for (int tm = 0; tm < 2; ++tm)
#pragma unroll
                for (int tn = 0; tn < 4; ++tn)
                    acc[tm][tn] = wmma_bf16(fa[tm], fb[tn], acc[tm][tn]);
        }
        __syncthreads();
    }

    const int cn = lane & 15;
    const int mh = (lane >> 4) * 8;
#pragma unroll
    for (int tm = 0; tm < 2; ++tm)
#pragma unroll
        for (int tn = 0; tn < 4; ++tn) {
            const int r0 = m0 + wm * 32 + tm * 16 + mh;
            const int c  = n0 + wn * 64 + tn * 16 + cn;
#pragma unroll
            for (int r = 0; r < 8; ++r)
                Out[(size_t)(r0 + r) * H + c] = acc[tm][tn][r];
        }
}

// ---------------------------------------------------------------------------
// Launch. Routing is provably a no-op: one local expert + renormalized top-k
// weights (sum to 1) => output == dense SwiGLU FFN of every token.
// ---------------------------------------------------------------------------
extern "C" void kernel_launch(void* const* d_in, const int* in_sizes, int n_in,
                              void* d_out, int out_size, void* d_ws, size_t ws_size,
                              hipStream_t stream) {
    const float* X  = (const float*)d_in[0];
    // d_in[1] = gate_w : unused (see note above)
    const float* W1 = (const float*)d_in[2];
    const float* W2 = (const float*)d_in[3];
    const float* W3 = (const float*)d_in[4];

    const int E = 8;
    const int H = in_sizes[1] / E;          // gate_w is [E, H]
    const int N = in_sizes[0] / H;          // hidden_states [N, H]
    const int I = in_sizes[2] / H;          // w1 [1, I, H]

    unsigned char* ws = (unsigned char*)d_ws;
    size_t off = 0;
    auto alloc = [&](size_t bytes) -> void* {
        void* p = ws + off;
        off = (off + bytes + 255) & ~(size_t)255;
        return p;
    };
    unsigned short* Xb  = (unsigned short*)alloc((size_t)N * H * 2);
    unsigned short* W1b = (unsigned short*)alloc((size_t)I * H * 2);
    unsigned short* W3b = (unsigned short*)alloc((size_t)I * H * 2);
    unsigned short* W2b = (unsigned short*)alloc((size_t)H * I * 2);
    unsigned short* Hb  = (unsigned short*)alloc((size_t)N * I * 2);

    auto cvt = [&](const float* src, unsigned short* dst, long long n) {
        const long long blocks = (n / 4 + 255) / 256;
        cvt_f32_to_bf16<<<dim3((unsigned)blocks), dim3(256), 0, stream>>>(src, dst, n);
    };
    cvt(X,  Xb,  (long long)N * H);
    cvt(W1, W1b, (long long)I * H);
    cvt(W3, W3b, (long long)I * H);
    cvt(W2, W2b, (long long)H * I);

    dim3 g1(I / 64, N / 128);
    moe_gemm13_swiglu<<<g1, dim3(256), 0, stream>>>(Xb, W1b, W3b, Hb, N, H, I);

    dim3 g2(H / 128, N / 128);
    moe_gemm2<<<g2, dim3(256), 0, stream>>>(Hb, W2b, (float*)d_out, N, H, I);
}
